// MoETransformerBlock_68728066670794
// MI455X (gfx1250) — matrix operand order repaired and checked by reading, not compiled, not run on hardware
//
#include <hip/hip_runtime.h>
#include <hip/hip_bf16.h>
#include <math.h>

// ---------------------------------------------------------------------------
// MoE transformer block forward for gfx1250 (MI455X), wave32 + WMMA bf16.
// Matrix ops: v_wmma_f32_16x16x32_bf16. GEMMs: 4-wave blocks, 64x64 tiles,
// B tile staged in LDS via GLOBAL_LOAD_ASYNC_TO_LDS_B128 (ASYNCcnt path).
// ---------------------------------------------------------------------------

typedef __attribute__((ext_vector_type(16))) __bf16 v16bf;
typedef __attribute__((ext_vector_type(8)))  float  v8f;

union BF16x16 {
  v16bf v;
  unsigned short u[16];
  uint4 q[2];
};

__device__ __forceinline__ unsigned short f2bf(float f) {
  union { float f; unsigned u; } t; t.f = f;
  unsigned b = t.u;
  unsigned r = b + 0x7FFFu + ((b >> 16) & 1u); // RNE
  return (unsigned short)(r >> 16);
}
__device__ __forceinline__ float bf2f(unsigned short s) {
  union { unsigned u; float f; } t; t.u = ((unsigned)s) << 16;
  return t.f;
}
__device__ __forceinline__ v8f wmma_bf16(v16bf a, v16bf b, v8f c) {
  return __builtin_amdgcn_wmma_f32_16x16x32_bf16(false, a, false, b, (short)0, c,
                                                 false, false);
}
__device__ __forceinline__ v8f zero8() {
  v8f z = {0.f, 0.f, 0.f, 0.f, 0.f, 0.f, 0.f, 0.f};
  return z;
}

// ---- async global->LDS staging (gfx1250 ASYNCcnt path, guarded) -----------
#if __has_builtin(__builtin_amdgcn_global_load_async_to_lds_b128) && \
    __has_builtin(__builtin_amdgcn_s_wait_asynccnt)
#define USE_ASYNC_LDS 1
typedef int v4i_async __attribute__((vector_size(16)));
typedef __attribute__((address_space(1))) v4i_async* g_v4p;
typedef __attribute__((address_space(3))) v4i_async* l_v4p;
#else
#define USE_ASYNC_LDS 0
#endif

__device__ __forceinline__ void lds_copy16(unsigned short* dst_lds,
                                           const unsigned short* src_glob) {
#if USE_ASYNC_LDS
  __builtin_amdgcn_global_load_async_to_lds_b128((g_v4p)src_glob, (l_v4p)dst_lds,
                                                 0, 0);
#else
  *(uint4*)dst_lds = *(const uint4*)src_glob;
#endif
}
__device__ __forceinline__ void async_fence() {
#if USE_ASYNC_LDS
  __builtin_amdgcn_s_wait_asynccnt(0);
#endif
}

#define BB   2
#define TT   1024
#define CCH  768
#define NNT  (BB * TT)        // 2048 tokens
#define NHD  12
#define NHK  4
#define HDD  64
#define CKV  (NHK * HDD)      // 256
#define NE   8
#define FFD  3072
#define PADROWS (2 * NNT + 16 * NE)  // 4224 gathered rows (16-padded per expert)
#define BPAD 40                      // LDS B-tile row pitch (ushorts): 80B, bank-spread

// ------------------------------- utility kernels ---------------------------

__global__ void init_kernel(int* cnt, float* S) {
  int t = threadIdx.x;
  if (t < NE) { cnt[t] = 0; S[t] = 0.f; }
}

__global__ void rmsnorm_kernel(const float* __restrict__ x,
                               const float* __restrict__ w,
                               unsigned short* __restrict__ outbf) {
  int row = blockIdx.x, tid = threadIdx.x;
  __shared__ float red[256];
  const float* xr = x + (size_t)row * CCH;
  float ss = 0.f;
  for (int c = tid; c < CCH; c += 256) { float v = xr[c]; ss += v * v; }
  red[tid] = ss;
  __syncthreads();
  for (int s = 128; s > 0; s >>= 1) {
    if (tid < s) red[tid] += red[tid + s];
    __syncthreads();
  }
  float sc = rsqrtf(red[0] / (float)CCH + 1e-6f);
  unsigned short* orow = outbf + (size_t)row * CCH;
  for (int c = tid; c < CCH; c += 256) orow[c] = f2bf(xr[c] * sc * w[c]);
}

// -------- batched transpose + fp32->bf16: in[e][K][M] -> out[e][M][K] ------
__global__ void transpose_cvt(const float* __restrict__ in,
                              unsigned short* __restrict__ outb,
                              int K, int M) {
  __shared__ float tile[32][33];
  int e = blockIdx.z;
  const float* inp = in + (size_t)e * K * M;
  unsigned short* outp = outb + (size_t)e * K * M;
  int m0 = blockIdx.x * 32, k0 = blockIdx.y * 32;
  int tx = threadIdx.x, ty = threadIdx.y;
#pragma unroll
  for (int i = 0; i < 32; i += 8)
    tile[ty + i][tx] = inp[(size_t)(k0 + ty + i) * M + m0 + tx];
  __syncthreads();
#pragma unroll
  for (int i = 0; i < 32; i += 8)
    outp[(size_t)(m0 + ty + i) * K + k0 + tx] = f2bf(tile[tx][ty + i]);
}

// ------------- WMMA GEMM: out[N x M](f32) = A[N x K](bf16) @ Wt ------------
// Wt bf16 [M][K]. Block = 4 waves, 64x64 tile; B staged in LDS, shared.
template <bool HASRES>
__global__ __launch_bounds__(128) void gemm_bf16t(
    const unsigned short* __restrict__ A, const unsigned short* __restrict__ Wt,
    float* __restrict__ out, const float* __restrict__ resid, int K, int M) {
  __shared__ unsigned short Bs[64][BPAD];
  int t = threadIdx.x;
  int w = t >> 5, l = t & 31;
  int n = l & 15, half = l >> 4, m = l & 15;
  int col0 = blockIdx.x * 64;
  int row0 = blockIdx.y * 64 + w * 16;
  v8f acc[4];
#pragma unroll
  for (int nt = 0; nt < 4; ++nt) acc[nt] = zero8();
  const unsigned short* arow = A + (size_t)(row0 + m) * K;
  int jj = t >> 1, ko = (t & 1) * 16;  // cooperative B loader: 32B per thread
  const unsigned short* bsrc = Wt + (size_t)(col0 + jj) * K + ko;
  for (int kb = 0; kb < K; kb += 32) {
    __builtin_prefetch(bsrc + kb + 32, 0, 1);  // global_prefetch next k-step
    __syncthreads();
    lds_copy16(&Bs[jj][ko],     bsrc + kb);
    lds_copy16(&Bs[jj][ko + 8], bsrc + kb + 8);
    async_fence();
    __syncthreads();
    BF16x16 a;
    a.q[0] = *(const uint4*)(arow + kb + half * 16);
    a.q[1] = *(const uint4*)(arow + kb + half * 16 + 8);
#pragma unroll
    for (int nt = 0; nt < 4; ++nt) {
      BF16x16 b;
      b.q[0] = *(const uint4*)&Bs[nt * 16 + n][half * 16];
      b.q[1] = *(const uint4*)&Bs[nt * 16 + n][half * 16 + 8];
      acc[nt] = wmma_bf16(a.v, b.v, acc[nt]);
    }
  }
#pragma unroll
  for (int r = 0; r < 8; ++r) {
    size_t rowoff = (size_t)(row0 + 8 * half + r) * M + col0 + n;
#pragma unroll
    for (int nt = 0; nt < 4; ++nt) {
      float v = acc[nt][r];
      if (HASRES) v += resid[rowoff + nt * 16];
      out[rowoff + nt * 16] = v;
    }
  }
}

// ----------------- RoPE + fp32->bf16 for q and k ---------------------------
__global__ void rope_cvt_kernel(const float* __restrict__ q,
                                const float* __restrict__ k,
                                unsigned short* __restrict__ qbf,
                                unsigned short* __restrict__ kbf,
                                const float* __restrict__ cosb,
                                const float* __restrict__ sinb) {
  int i = blockIdx.x * blockDim.x + threadIdx.x;
  const int qtot = NNT * NHD * (HDD / 2);
  const int ktot = NNT * NHK * (HDD / 2);
  if (i < qtot) {
    int d = i & 31, rest = i >> 5;
    int h = rest % NHD, nrow = rest / NHD, t = nrow % TT;
    float c0 = cosb[t * HDD + d], s0 = sinb[t * HDD + d];
    float c1 = cosb[t * HDD + 32 + d], s1 = sinb[t * HDD + 32 + d];
    const float* p = q + (size_t)nrow * CCH + h * HDD;
    unsigned short* o = qbf + (size_t)nrow * CCH + h * HDD;
    float a = p[d], b = p[d + 32];
    o[d]      = f2bf(a * c0 - b * s0);
    o[d + 32] = f2bf(b * c1 + a * s1);
  } else if (i < qtot + ktot) {
    int j = i - qtot;
    int d = j & 31, rest = j >> 5;
    int h = rest % NHK, nrow = rest / NHK, t = nrow % TT;
    float c0 = cosb[t * HDD + d], s0 = sinb[t * HDD + d];
    float c1 = cosb[t * HDD + 32 + d], s1 = sinb[t * HDD + 32 + d];
    const float* p = k + (size_t)nrow * CKV + h * HDD;
    unsigned short* o = kbf + (size_t)nrow * CKV + h * HDD;
    float a = p[d], b = p[d + 32];
    o[d]      = f2bf(a * c0 - b * s0);
    o[d + 32] = f2bf(b * c1 + a * s1);
  }
}

// -------- V: [B,T,Hk,hd] fp32 -> [B,Hk,hd,T] bf16 (B-operand friendly) -----
__global__ void vtrans_kernel(const float* __restrict__ v,
                              unsigned short* __restrict__ vtb) {
  int i = blockIdx.x * blockDim.x + threadIdx.x;
  if (i >= NNT * CKV) return;  // i = ((b*Hk + kh)*hd + d)*T + t
  int t = i % TT, rest = i / TT;
  int d = rest % HDD, rest2 = rest / HDD;
  int kh = rest2 % NHK, b = rest2 / NHK;
  vtb[i] = f2bf(v[((size_t)b * TT + t) * CKV + kh * HDD + d]);
}

// ----------------- flash attention, one wave per (b,h,16 q rows) -----------
__global__ void attn_kernel(const unsigned short* __restrict__ qbf,
                            const unsigned short* __restrict__ kbf,
                            const unsigned short* __restrict__ vtb,
                            unsigned short* __restrict__ obf) {
  int qt = blockIdx.x, hh = blockIdx.y, b = blockIdx.z;
  int kh = hh / (NHD / NHK);
  int l = threadIdx.x;
  int n = l & 15, half = l >> 4, m = l & 15;

  BF16x16 qa0, qa1;  // Q as A-operand, two K=32 steps over hd=64
  {
    const unsigned short* qr =
        qbf + ((size_t)b * TT + qt * 16 + m) * CCH + hh * HDD;
    qa0.q[0] = *(const uint4*)(qr + half * 16);
    qa0.q[1] = *(const uint4*)(qr + half * 16 + 8);
    qa1.q[0] = *(const uint4*)(qr + 32 + half * 16);
    qa1.q[1] = *(const uint4*)(qr + 40 + half * 16);
  }
  const float scale = 0.125f;  // 1/sqrt(64)
  v8f oacc[4];
#pragma unroll
  for (int nt = 0; nt < 4; ++nt) oacc[nt] = zero8();
  float mrow[8], lrow[8];
#pragma unroll
  for (int r = 0; r < 8; ++r) { mrow[r] = -3.0e38f; lrow[r] = 0.f; }
  __shared__ float Pbuf[16][33];
  int qhi = qt * 16 + 15;

  for (int k0 = 0; k0 <= qhi; k0 += 32) {
    v8f sa = zero8(), sb = zero8();
    {  // score tile a: keys k0..k0+15
      BF16x16 kf;
      const unsigned short* kr =
          kbf + ((size_t)b * TT + k0 + n) * CKV + kh * HDD + half * 16;
      kf.q[0] = *(const uint4*)kr;
      kf.q[1] = *(const uint4*)(kr + 8);
      sa = wmma_bf16(qa0.v, kf.v, sa);
      kf.q[0] = *(const uint4*)(kr + 32);
      kf.q[1] = *(const uint4*)(kr + 40);
      sa = wmma_bf16(qa1.v, kf.v, sa);
    }
    {  // score tile b: keys k0+16..k0+31
      BF16x16 kf;
      const unsigned short* kr =
          kbf + ((size_t)b * TT + k0 + 16 + n) * CKV + kh * HDD + half * 16;
      kf.q[0] = *(const uint4*)kr;
      kf.q[1] = *(const uint4*)(kr + 8);
      sb = wmma_bf16(qa0.v, kf.v, sb);
      kf.q[0] = *(const uint4*)(kr + 32);
      kf.q[1] = *(const uint4*)(kr + 40);
      sb = wmma_bf16(qa1.v, kf.v, sb);
    }
#pragma unroll
    for (int r = 0; r < 8; ++r) {  // online softmax (fp32)
      int qi = qt * 16 + 8 * half + r;
      float va  = (k0 + n      <= qi) ? sa[r] * scale : -3.0e38f;
      float vb2 = (k0 + 16 + n <= qi) ? sb[r] * scale : -3.0e38f;
      float mx = fmaxf(va, vb2);
#pragma unroll
      for (int s2 = 1; s2 < 16; s2 <<= 1) mx = fmaxf(mx, __shfl_xor(mx, s2, 32));
      float mnew  = fmaxf(mrow[r], mx);
      float alpha = __expf(mrow[r] - mnew);
      float pa = __expf(va - mnew);
      float pb = __expf(vb2 - mnew);
      float rs = pa + pb;
#pragma unroll
      for (int s2 = 1; s2 < 16; s2 <<= 1) rs += __shfl_xor(rs, s2, 32);
      lrow[r] = lrow[r] * alpha + rs;
      mrow[r] = mnew;
#pragma unroll
      for (int nt = 0; nt < 4; ++nt) oacc[nt][r] *= alpha;
      Pbuf[8 * half + r][n]      = pa;
      Pbuf[8 * half + r][16 + n] = pb;
    }
    __syncthreads();
    BF16x16 pf;  // P (16x32) restaged from D-layout to A-layout via LDS
#pragma unroll
    for (int i = 0; i < 8; ++i) {
      pf.u[i]     = f2bf(Pbuf[m][half * 16 + i]);
      pf.u[8 + i] = f2bf(Pbuf[m][half * 16 + 8 + i]);
    }
    __syncthreads();
#pragma unroll
    for (int nt = 0; nt < 4; ++nt) {  // o += P @ V, 4 d-tiles of 16
      BF16x16 vf;
      const unsigned short* vr =
          vtb + ((size_t)(b * NHK + kh) * HDD + nt * 16 + n) * TT + k0 + half * 16;
      vf.q[0] = *(const uint4*)vr;
      vf.q[1] = *(const uint4*)(vr + 8);
      oacc[nt] = wmma_bf16(pf.v, vf.v, oacc[nt]);
    }
  }
#pragma unroll
  for (int r = 0; r < 8; ++r) {
    float inv = 1.f / lrow[r];
    int t = qt * 16 + 8 * half + r;
    unsigned short* orow = obf + ((size_t)b * TT + t) * CCH + hh * HDD;
#pragma unroll
    for (int nt = 0; nt < 4; ++nt) orow[nt * 16 + n] = f2bf(oacc[nt][r] * inv);
  }
}

// ------------------------------ router -------------------------------------
__global__ void logits_kernel(const unsigned short* __restrict__ h2bf,
                              const float* __restrict__ Wg,
                              float* __restrict__ logits) {
  int i = blockIdx.x * blockDim.x + threadIdx.x;
  if (i >= NNT * NE) return;
  int e = i & 7, nrow = i >> 3;
  const unsigned short* hr = h2bf + (size_t)nrow * CCH;
  float s = 0.f;
  for (int c = 0; c < CCH; ++c) s += bf2f(hr[c]) * Wg[(size_t)c * NE + e];
  logits[i] = s;
}

__global__ void gate_topk(const float* __restrict__ logits,
                          float* __restrict__ topw, int* __restrict__ tix,
                          int* __restrict__ cnt) {
  int nrow = blockIdx.x * blockDim.x + threadIdx.x;
  if (nrow >= NNT) return;
  float lg[NE];
#pragma unroll
  for (int e = 0; e < NE; ++e) lg[e] = logits[nrow * NE + e];
  int b0 = 0; float v0 = lg[0];
#pragma unroll
  for (int e = 1; e < NE; ++e) if (lg[e] > v0) { v0 = lg[e]; b0 = e; }
  int b1 = (b0 == 0) ? 1 : 0; float v1 = lg[b1];
#pragma unroll
  for (int e = 0; e < NE; ++e)
    if (e != b0 && lg[e] > v1) { v1 = lg[e]; b1 = e; }
  float w0 = 1.f / (1.f + __expf(v1 - v0));
  topw[nrow * 2] = w0; topw[nrow * 2 + 1] = 1.f - w0;
  tix[nrow * 2] = b0;  tix[nrow * 2 + 1] = b1;
  atomicAdd(&cnt[b0], 1);
  atomicAdd(&cnt[b1], 1);
}

__global__ void offsets_kernel(const int* __restrict__ cnt, int* __restrict__ off,
                               int* __restrict__ fill) {
  if (threadIdx.x == 0 && blockIdx.x == 0) {
    int acc = 0;
    for (int e = 0; e < NE; ++e) {
      off[e] = acc;
      acc += ((cnt[e] + 15) >> 4) << 4;  // 16-row padded segments
      fill[e] = 0;
    }
    off[NE] = acc;
  }
}

__global__ void assign_kernel(const int* __restrict__ tix,
                              const int* __restrict__ off, int* __restrict__ fill,
                              int* __restrict__ tok, int* __restrict__ slotOf) {
  int nrow = blockIdx.x * blockDim.x + threadIdx.x;
  if (nrow >= NNT) return;
  for (int kk = 0; kk < 2; ++kk) {
    int e = tix[nrow * 2 + kk];
    int j = atomicAdd(&fill[e], 1);
    int slot = off[e] + j;
    tok[slot] = nrow;
    slotOf[nrow * 2 + kk] = slot;
  }
}

// ------------- gathered expert FFN: g = silu(x@W1) * (x@W3)  (bf16) --------
__global__ __launch_bounds__(128) void moe_h13(
    const unsigned short* __restrict__ h2bf,
    const unsigned short* __restrict__ w1t, const unsigned short* __restrict__ w3t,
    const int* __restrict__ tok, const int* __restrict__ cnt,
    const int* __restrict__ off, unsigned short* __restrict__ g) {
  __shared__ unsigned short B1s[64][BPAD];
  __shared__ unsigned short B3s[64][BPAD];
  int e = blockIdx.z;
  int count = cnt[e];
  int t0 = blockIdx.y * 64;
  if (t0 >= count) return;  // uniform over block
  int count16 = (count + 15) & ~15;
  int c0 = blockIdx.x * 64;
  int base = off[e];
  int t = threadIdx.x;
  int w = t >> 5, l = t & 31;
  int n = l & 15, half = l >> 4, m = l & 15;
  int rt = t0 + w * 16;                 // this wave's row tile
  bool active = rt < count16;           // whether results are stored
  int ridx = rt + m; if (ridx >= count) ridx = count - 1;  // clamp: loads stay valid
  int token = tok[base + ridx];
  const unsigned short* arow = h2bf + (size_t)token * CCH;
  const unsigned short* w1e = w1t + (size_t)e * FFD * CCH;
  const unsigned short* w3e = w3t + (size_t)e * FFD * CCH;
  v8f a1[4], a3[4];
#pragma unroll
  for (int nt = 0; nt < 4; ++nt) { a1[nt] = zero8(); a3[nt] = zero8(); }
  int jj = t >> 1, ko = (t & 1) * 16;
  const unsigned short* b1src = w1e + (size_t)(c0 + jj) * CCH + ko;
  const unsigned short* b3src = w3e + (size_t)(c0 + jj) * CCH + ko;
  for (int kb = 0; kb < CCH; kb += 32) {
    __builtin_prefetch(b1src + kb + 32, 0, 1);
    __builtin_prefetch(b3src + kb + 32, 0, 1);
    __syncthreads();
    lds_copy16(&B1s[jj][ko],     b1src + kb);
    lds_copy16(&B1s[jj][ko + 8], b1src + kb + 8);
    lds_copy16(&B3s[jj][ko],     b3src + kb);
    lds_copy16(&B3s[jj][ko + 8], b3src + kb + 8);
    async_fence();
    __syncthreads();
    BF16x16 a;
    a.q[0] = *(const uint4*)(arow + kb + half * 16);
    a.q[1] = *(const uint4*)(arow + kb + half * 16 + 8);
#pragma unroll
    for (int nt = 0; nt < 4; ++nt) {
      BF16x16 b1, b3;
      b1.q[0] = *(const uint4*)&B1s[nt * 16 + n][half * 16];
      b1.q[1] = *(const uint4*)&B1s[nt * 16 + n][half * 16 + 8];
      b3.q[0] = *(const uint4*)&B3s[nt * 16 + n][half * 16];
      b3.q[1] = *(const uint4*)&B3s[nt * 16 + n][half * 16 + 8];
      a1[nt] = wmma_bf16(a.v, b1.v, a1[nt]);
      a3[nt] = wmma_bf16(a.v, b3.v, a3[nt]);
    }
  }
  if (active) {
#pragma unroll
    for (int r = 0; r < 8; ++r) {
      size_t rowoff = (size_t)(base + rt + 8 * half + r) * FFD + c0 + n;
#pragma unroll
      for (int nt = 0; nt < 4; ++nt) {
        float h1v = a1[nt][r];
        float sil = h1v / (1.f + __expf(-h1v));
        g[rowoff + nt * 16] = f2bf(sil * a3[nt][r]);
      }
    }
  }
}

// ----------------------- y = g @ W2 (per expert, gathered) -----------------
__global__ __launch_bounds__(128) void moe_y(
    const unsigned short* __restrict__ g, const unsigned short* __restrict__ w2t,
    const int* __restrict__ cnt, const int* __restrict__ off,
    float* __restrict__ y) {
  __shared__ unsigned short Bs[64][BPAD];
  int e = blockIdx.z;
  int count = cnt[e];
  int t0 = blockIdx.y * 64;
  if (t0 >= count) return;  // uniform over block
  int count16 = (count + 15) & ~15;
  int c0 = blockIdx.x * 64;
  int base = off[e];
  int t = threadIdx.x;
  int w = t >> 5, l = t & 31;
  int n = l & 15, half = l >> 4, m = l & 15;
  int rt = t0 + w * 16;
  bool active = rt < count16;
  int ridx = rt + m; if (ridx >= count16) ridx = count16 - 1;  // stay in padded seg
  const unsigned short* arow = g + (size_t)(base + ridx) * FFD;
  const unsigned short* w2e = w2t + (size_t)e * CCH * FFD;
  v8f acc[4];
#pragma unroll
  for (int nt = 0; nt < 4; ++nt) acc[nt] = zero8();
  int jj = t >> 1, ko = (t & 1) * 16;
  const unsigned short* bsrc = w2e + (size_t)(c0 + jj) * FFD + ko;
  for (int kb = 0; kb < FFD; kb += 32) {
    __builtin_prefetch(bsrc + kb + 32, 0, 1);
    __syncthreads();
    lds_copy16(&Bs[jj][ko],     bsrc + kb);
    lds_copy16(&Bs[jj][ko + 8], bsrc + kb + 8);
    async_fence();
    __syncthreads();
    BF16x16 a;
    a.q[0] = *(const uint4*)(arow + kb + half * 16);
    a.q[1] = *(const uint4*)(arow + kb + half * 16 + 8);
#pragma unroll
    for (int nt = 0; nt < 4; ++nt) {
      BF16x16 b;
      b.q[0] = *(const uint4*)&Bs[nt * 16 + n][half * 16];
      b.q[1] = *(const uint4*)&Bs[nt * 16 + n][half * 16 + 8];
      acc[nt] = wmma_bf16(a.v, b.v, acc[nt]);
    }
  }
  if (active) {
#pragma unroll
    for (int r = 0; r < 8; ++r) {
      size_t rowoff = (size_t)(base + rt + 8 * half + r) * CCH + c0 + n;
#pragma unroll
      for (int nt = 0; nt < 4; ++nt) y[rowoff + nt * 16] = acc[nt][r];
    }
  }
}

// ----------- deterministic combine: out += w0*y[s0] + w1*y[s1] -------------
__global__ void combine_kernel(const float* __restrict__ y,
                               const float* __restrict__ topw,
                               const int* __restrict__ slotOf,
                               float* __restrict__ out) {
  int i = blockIdx.x * blockDim.x + threadIdx.x;
  if (i >= NNT * CCH) return;
  int nrow = i / CCH, c = i - nrow * CCH;
  int s0 = slotOf[nrow * 2], s1 = slotOf[nrow * 2 + 1];
  out[i] += topw[nrow * 2]     * y[(size_t)s0 * CCH + c] +
            topw[nrow * 2 + 1] * y[(size_t)s1 * CCH + c];
}

// ------------------------------- aux loss ----------------------------------
__global__ void aux_reduce(const float* __restrict__ logits, float* __restrict__ S) {
  int e = blockIdx.x, tid = threadIdx.x;
  __shared__ float red[256];
  float part = 0.f;
  for (int nrow = tid; nrow < NNT; nrow += 256) {
    float mx = logits[nrow * NE];
    for (int j = 1; j < NE; ++j) mx = fmaxf(mx, logits[nrow * NE + j]);
    float den = 0.f;
    for (int j = 0; j < NE; ++j) den += __expf(logits[nrow * NE + j] - mx);
    part += __expf(logits[nrow * NE + e] - mx) / den;
  }
  red[tid] = part;
  __syncthreads();
  for (int s = 128; s > 0; s >>= 1) {
    if (tid < s) red[tid] += red[tid + s];
    __syncthreads();
  }
  if (tid == 0) S[e] = red[0];
}

__global__ void aux_final(const float* __restrict__ S, float* __restrict__ outAux) {
  if (threadIdx.x == 0 && blockIdx.x == 0) {
    float a = 0.f;
    for (int e = 0; e < NE; ++e) { float u = S[e] / (float)NNT; a += u * u; }
    *outAux = (float)NE * a;   // aux = E*sum((S/N)^2)
  }
}

// ---------------------------------------------------------------------------
extern "C" void kernel_launch(void* const* d_in, const int* in_sizes, int n_in,
                              void* d_out, int out_size, void* d_ws, size_t ws_size,
                              hipStream_t stream) {
  const float* x    = (const float*)d_in[0];
  const float* cosp = (const float*)d_in[1];
  const float* sinp = (const float*)d_in[2];
  const float* ln1  = (const float*)d_in[3];
  const float* ln2  = (const float*)d_in[4];
  const float* Wq   = (const float*)d_in[5];
  const float* Wk   = (const float*)d_in[6];
  const float* Wv   = (const float*)d_in[7];
  const float* Wo   = (const float*)d_in[8];
  const float* Wg   = (const float*)d_in[9];
  const float* W1   = (const float*)d_in[10];
  const float* W2   = (const float*)d_in[11];
  const float* W3   = (const float*)d_in[12];
  float* out = (float*)d_out;
  char* ws = (char*)d_ws;

  size_t cur = 0;
  auto take = [&](size_t bytes) {
    size_t p = cur;
    cur += (bytes + 255) & ~(size_t)255;
    return p;
  };
  unsigned short* h1bf = (unsigned short*)(ws + take((size_t)NNT * CCH * 2));
  float* qbuf   = (float*)(ws + take((size_t)NNT * CCH * 4));
  float* kbuf   = (float*)(ws + take((size_t)NNT * CKV * 4));
  float* vbuf   = (float*)(ws + take((size_t)NNT * CKV * 4));
  unsigned short* qbf = (unsigned short*)(ws + take((size_t)NNT * CCH * 2));
  unsigned short* kbf = (unsigned short*)(ws + take((size_t)NNT * CKV * 2));
  unsigned short* vtb = (unsigned short*)(ws + take((size_t)NNT * CKV * 2));
  unsigned short* obf  = (unsigned short*)(ws + take((size_t)NNT * CCH * 2));
  unsigned short* h2bf = (unsigned short*)(ws + take((size_t)NNT * CCH * 2));
  float* logits = (float*)(ws + take((size_t)NNT * NE * 4));
  float* topw   = (float*)(ws + take((size_t)NNT * 2 * 4));
  int*   tix    = (int*)(ws + take((size_t)NNT * 2 * 4));
  int*   slotOf = (int*)(ws + take((size_t)NNT * 2 * 4));
  int*   cnt    = (int*)(ws + take(NE * 4));
  int*   offE   = (int*)(ws + take((NE + 1) * 4));
  int*   fill   = (int*)(ws + take(NE * 4));
  float* S      = (float*)(ws + take(NE * 4));
  int*   tok    = (int*)(ws + take((size_t)PADROWS * 4));
  unsigned short* g = (unsigned short*)(ws + take((size_t)PADROWS * FFD * 2));
  float* ybuf   = (float*)(ws + take((size_t)PADROWS * CCH * 4));
  // bf16 transposed weight mirrors
  unsigned short* wqT = (unsigned short*)(ws + take((size_t)CCH * CCH * 2));
  unsigned short* wkT = (unsigned short*)(ws + take((size_t)CCH * CKV * 2));
  unsigned short* wvT = (unsigned short*)(ws + take((size_t)CCH * CKV * 2));
  unsigned short* woT = (unsigned short*)(ws + take((size_t)CCH * CCH * 2));
  unsigned short* w1T = (unsigned short*)(ws + take((size_t)NE * CCH * FFD * 2));
  unsigned short* w3T = (unsigned short*)(ws + take((size_t)NE * CCH * FFD * 2));
  unsigned short* w2T = (unsigned short*)(ws + take((size_t)NE * FFD * CCH * 2));
  (void)in_sizes; (void)n_in; (void)out_size; (void)ws_size;

  dim3 tb(32, 8);
  init_kernel<<<1, 32, 0, stream>>>(cnt, S);

  // one-time weight transpose + bf16 conversion (HBM bound, ~15us total)
  transpose_cvt<<<dim3(CCH / 32, CCH / 32, 1), tb, 0, stream>>>(Wq, wqT, CCH, CCH);
  transpose_cvt<<<dim3(CKV / 32, CCH / 32, 1), tb, 0, stream>>>(Wk, wkT, CCH, CKV);
  transpose_cvt<<<dim3(CKV / 32, CCH / 32, 1), tb, 0, stream>>>(Wv, wvT, CCH, CKV);
  transpose_cvt<<<dim3(CCH / 32, CCH / 32, 1), tb, 0, stream>>>(Wo, woT, CCH, CCH);
  transpose_cvt<<<dim3(FFD / 32, CCH / 32, NE), tb, 0, stream>>>(W1, w1T, CCH, FFD);
  transpose_cvt<<<dim3(FFD / 32, CCH / 32, NE), tb, 0, stream>>>(W3, w3T, CCH, FFD);
  transpose_cvt<<<dim3(CCH / 32, FFD / 32, NE), tb, 0, stream>>>(W2, w2T, FFD, CCH);

  // h = rmsnorm(x) -> bf16
  rmsnorm_kernel<<<NNT, 256, 0, stream>>>(x, ln1, h1bf);

  // q/k/v projections (WMMA bf16, 64x64 blocks, LDS-shared B)
  gemm_bf16t<false><<<dim3(CCH / 64, NNT / 64), 128, 0, stream>>>(h1bf, wqT, qbuf, nullptr, CCH, CCH);
  gemm_bf16t<false><<<dim3(CKV / 64, NNT / 64), 128, 0, stream>>>(h1bf, wkT, kbuf, nullptr, CCH, CKV);
  gemm_bf16t<false><<<dim3(CKV / 64, NNT / 64), 128, 0, stream>>>(h1bf, wvT, vbuf, nullptr, CCH, CKV);

  // RoPE -> bf16 q/k; V -> transposed bf16
  {
    int total = NNT * (NHD + NHK) * (HDD / 2);
    rope_cvt_kernel<<<(total + 255) / 256, 256, 0, stream>>>(qbuf, kbuf, qbf, kbf, cosp, sinp);
    vtrans_kernel<<<(NNT * CKV + 255) / 256, 256, 0, stream>>>(vbuf, vtb);
  }

  // causal GQA flash attention -> obf (bf16, [B,T,C] layout)
  attn_kernel<<<dim3(TT / 16, NHD, BB), 32, 0, stream>>>(qbf, kbf, vtb, obf);

  // x1 = x + o @ Wo  (written straight into d_out)
  gemm_bf16t<true><<<dim3(CCH / 64, NNT / 64), 128, 0, stream>>>(obf, woT, out, x, CCH, CCH);

  // h2 = rmsnorm(x1) -> bf16
  rmsnorm_kernel<<<NNT, 256, 0, stream>>>(out, ln2, h2bf);

  // router
  logits_kernel<<<(NNT * NE + 255) / 256, 256, 0, stream>>>(h2bf, Wg, logits);
  gate_topk<<<(NNT + 255) / 256, 256, 0, stream>>>(logits, topw, tix, cnt);
  offsets_kernel<<<1, 32, 0, stream>>>(cnt, offE, fill);
  assign_kernel<<<(NNT + 255) / 256, 256, 0, stream>>>(tix, offE, fill, tok, slotOf);

  // top-2 gathered expert FFN (WMMA bf16, 4-wave blocks, async-LDS B)
  moe_h13<<<dim3(FFD / 64, NNT / 64, NE), 128, 0, stream>>>(h2bf, w1T, w3T, tok, cnt, offE, g);
  moe_y<<<dim3(CCH / 64, NNT / 64, NE), 128, 0, stream>>>(g, w2T, cnt, offE, ybuf);

  // out = x1 + gate-weighted expert outputs (deterministic, no FP atomics)
  combine_kernel<<<(NNT * CCH + 255) / 256, 256, 0, stream>>>(ybuf, topw, slotOf, out);

  // aux loss
  aux_reduce<<<NE, 256, 0, stream>>>(logits, S);
  aux_final<<<1, 32, 0, stream>>>(S, out + (size_t)NNT * CCH);
}
